// QuantizedLinear_41910290874849
// MI455X (gfx1250) — 8-bit matrix operand order repaired and checked by reading, compiled, not run on hardware
//
#include <hip/hip_runtime.h>
#include <hip/hip_bf16.h>

#define RADIX 4
#define NUM_LAYERS 3
#define IN 1024
#define OUT 1024
#define BM 128
#define BN 128
#define BK 64

typedef __attribute__((ext_vector_type(8))) int v8i;
typedef __attribute__((ext_vector_type(4))) int v4i;

#if __has_builtin(__builtin_amdgcn_global_load_async_to_lds_b128)
#define HAS_ASYNC_LDS 1
#else
#define HAS_ASYNC_LDS 0
#endif

__device__ __forceinline__ void async_cp16(void* lds, const void* g) {
#if HAS_ASYNC_LDS
    __builtin_amdgcn_global_load_async_to_lds_b128(
        (__attribute__((address_space(1))) v4i*)(void*)g,
        (__attribute__((address_space(3))) v4i*)lds,
         0,  0);
#else
    (void)lds; (void)g;
#endif
}

__device__ __forceinline__ void wait_async0() {
#if __has_builtin(__builtin_amdgcn_s_wait_asynccnt)
    __builtin_amdgcn_s_wait_asynccnt(0);
#else
    asm volatile("s_wait_asynccnt 0x0" ::: "memory");
#endif
}

__device__ __forceinline__ int lattice_idx(float v, float s0) {
    float r = v;
    int k = 0;
#pragma unroll
    for (int i = NUM_LAYERS - 1; i >= 0; --i) {
        const float s = s0 * (float)(1 << i);
        float fidx = rintf(r / s);
        fidx = fminf(fmaxf(fidx, -(float)RADIX), (float)RADIX);
        r -= fidx * s;
        k += ((int)fidx) << i;
    }
    return k;
}

__global__ void lattice_quant_kernel(const float* __restrict__ in,
                                     signed char* __restrict__ out,
                                     const float* __restrict__ scales,
                                     int n4) {
    const int i = blockIdx.x * blockDim.x + threadIdx.x;
    if (i >= n4) return;
    const float s0 = scales[0];
    const float4 v = reinterpret_cast<const float4*>(in)[i];
    const int k0 = lattice_idx(v.x, s0) & 0xff;
    const int k1 = lattice_idx(v.y, s0) & 0xff;
    const int k2 = lattice_idx(v.z, s0) & 0xff;
    const int k3 = lattice_idx(v.w, s0) & 0xff;
    reinterpret_cast<int*>(out)[i] = k0 | (k1 << 8) | (k2 << 16) | (k3 << 24);
}

__global__ __launch_bounds__(256)
void iu8_gemm_kernel(const signed char* __restrict__ Xq,
                     const signed char* __restrict__ Wq,
                     const float* __restrict__ bias,
                     const float* __restrict__ scales,
                     float* __restrict__ out) {
    __shared__ __align__(16) signed char lA[2][BM * BK];
    __shared__ __align__(16) signed char lB[2][BN * BK];

    const int tid   = threadIdx.x;
    const int lane  = tid & 31;
    const int wave  = tid >> 5;
    const int l15   = lane & 15;
    const int lhalf = lane >> 4;

    const int wm = wave & 3;
    const int wn = wave >> 2;

    const int bm = blockIdx.x * BM;
    const int bn = blockIdx.y * BN;

    const signed char* gA = Xq + (size_t)bm * IN;
    const signed char* gB = Wq + (size_t)bn * IN;

    const int ch0 = tid;
    const int ch1 = tid + 256;
    const int rowA0 = ch0 >> 2, colA0 = (ch0 & 3) * 16;
    const int rowA1 = ch1 >> 2, colA1 = (ch1 & 3) * 16;

    v8i acc[2][4];
#pragma unroll
    for (int mt = 0; mt < 2; ++mt)
#pragma unroll
        for (int nt = 0; nt < 4; ++nt)
#pragma unroll
            for (int r = 0; r < 8; ++r) acc[mt][nt][r] = 0;

#if HAS_ASYNC_LDS
    async_cp16(lA[0] + ch0 * 16, gA + (size_t)rowA0 * IN + colA0);
    async_cp16(lA[0] + ch1 * 16, gA + (size_t)rowA1 * IN + colA1);
    async_cp16(lB[0] + ch0 * 16, gB + (size_t)rowA0 * IN + colA0);
    async_cp16(lB[0] + ch1 * 16, gB + (size_t)rowA1 * IN + colA1);
    wait_async0();
#else
    {
        const int4 a0 = *reinterpret_cast<const int4*>(gA + (size_t)rowA0 * IN + colA0);
        const int4 a1 = *reinterpret_cast<const int4*>(gA + (size_t)rowA1 * IN + colA1);
        const int4 b0 = *reinterpret_cast<const int4*>(gB + (size_t)rowA0 * IN + colA0);
        const int4 b1 = *reinterpret_cast<const int4*>(gB + (size_t)rowA1 * IN + colA1);
        reinterpret_cast<int4*>(lA[0])[ch0] = a0;
        reinterpret_cast<int4*>(lA[0])[ch1] = a1;
        reinterpret_cast<int4*>(lB[0])[ch0] = b0;
        reinterpret_cast<int4*>(lB[0])[ch1] = b1;
    }
#endif
    __syncthreads();

    int buf = 0;
    for (int kk = 0; kk < IN; kk += BK) {
        const bool has_next = (kk + BK) < IN;

#if HAS_ASYNC_LDS
        if (has_next) {
            const int nk = kk + BK;
            signed char* nxtA = lA[buf ^ 1];
            signed char* nxtB = lB[buf ^ 1];
            async_cp16(nxtA + ch0 * 16, gA + (size_t)rowA0 * IN + nk + colA0);
            async_cp16(nxtA + ch1 * 16, gA + (size_t)rowA1 * IN + nk + colA1);
            async_cp16(nxtB + ch0 * 16, gB + (size_t)rowA0 * IN + nk + colA0);
            async_cp16(nxtB + ch1 * 16, gB + (size_t)rowA1 * IN + nk + colA1);
        }
#else
        int4 na0, na1, nb0, nb1;
        if (has_next) {
            const int nk = kk + BK;
            na0 = *reinterpret_cast<const int4*>(gA + (size_t)rowA0 * IN + nk + colA0);
            na1 = *reinterpret_cast<const int4*>(gA + (size_t)rowA1 * IN + nk + colA1);
            nb0 = *reinterpret_cast<const int4*>(gB + (size_t)rowA0 * IN + nk + colA0);
            nb1 = *reinterpret_cast<const int4*>(gB + (size_t)rowA1 * IN + nk + colA1);
        }
#endif
        if (kk + 2 * BK < IN) {
            const int pk = kk + 2 * BK;
            __builtin_prefetch(gA + (size_t)(tid >> 1) * IN + pk + (tid & 1) * 32, 0, 1);
            __builtin_prefetch(gB + (size_t)(tid >> 1) * IN + pk + (tid & 1) * 32, 0, 1);
        }

        const signed char* curA = lA[buf];
        const signed char* curB = lB[buf];

        v8i afrag[2];
#pragma unroll
        for (int mt = 0; mt < 2; ++mt) {
            const signed char* base = curA + (wm * 32 + mt * 16 + l15) * BK;
#pragma unroll
            for (int v = 0; v < 8; ++v) {
                const int kb = (v >> 1) * 16 + (v & 1) * 4 + lhalf * 8;
                afrag[mt][v] = *reinterpret_cast<const int*>(base + kb);
            }
        }
        v8i bfrag[4];
#pragma unroll
        for (int nt = 0; nt < 4; ++nt) {
            const signed char* base = curB + (wn * 64 + nt * 16 + l15) * BK;
#pragma unroll
            for (int v = 0; v < 8; ++v) {
                const int kb = (v >> 1) * 16 + (v & 1) * 4 + lhalf * 8;
                bfrag[nt][v] = *reinterpret_cast<const int*>(base + kb);
            }
        }
#pragma unroll
        for (int mt = 0; mt < 2; ++mt)
#pragma unroll
            for (int nt = 0; nt < 4; ++nt)
                acc[mt][nt] = __builtin_amdgcn_wmma_i32_16x16x64_iu8(
                     true, afrag[mt],
                     true, bfrag[nt],
                    acc[mt][nt],
                     false,  false);

        if (has_next) {
#if HAS_ASYNC_LDS
            wait_async0();
#else
            signed char* nxtA = lA[buf ^ 1];
            signed char* nxtB = lB[buf ^ 1];
            reinterpret_cast<int4*>(nxtA)[ch0] = na0;
            reinterpret_cast<int4*>(nxtA)[ch1] = na1;
            reinterpret_cast<int4*>(nxtB)[ch0] = nb0;
            reinterpret_cast<int4*>(nxtB)[ch1] = nb1;
#endif
            __syncthreads();
        }
        buf ^= 1;
    }

    const float s0 = scales[0];
    const float s2 = s0 * s0;
#pragma unroll
    for (int mt = 0; mt < 2; ++mt) {
#pragma unroll
        for (int nt = 0; nt < 4; ++nt) {
            const int col  = bn + wn * 64 + nt * 16 + l15;
            const int row0 = bm + wm * 32 + mt * 16 + lhalf * 8;
            const float bb = bias[col];
#pragma unroll
            for (int r = 0; r < 8; ++r) {
                out[(size_t)(row0 + r) * OUT + col] =
                    s2 * (float)acc[mt][nt][r] + bb;
            }
        }
    }
}

extern "C" void kernel_launch(void* const* d_in, const int* in_sizes, int n_in,
                              void* d_out, int out_size, void* d_ws, size_t ws_size,
                              hipStream_t stream) {
    const float* x      = (const float*)d_in[0];
    const float* weight = (const float*)d_in[1];
    const float* bias   = (const float*)d_in[2];
    const float* scales = (const float*)d_in[3];
    float* out = (float*)d_out;

    const int nx = in_sizes[0];
    const int nw = in_sizes[1];
    const int B  = nx / IN;

    signed char* xq = (signed char*)d_ws;
    signed char* wq = xq + (size_t)nx;

    {
        const int n4 = nx / 4;
        lattice_quant_kernel<<<(n4 + 255) / 256, 256, 0, stream>>>(x, xq, scales, n4);
    }
    {
        const int n4 = nw / 4;
        lattice_quant_kernel<<<(n4 + 255) / 256, 256, 0, stream>>>(weight, wq, scales, n4);
    }

    dim3 grid(B / BM, OUT / BN);
    iu8_gemm_kernel<<<grid, 256, 0, stream>>>(xq, wq, bias, scales, out);
}
